// CompressedSparseAttention_40286793236980
// MI455X (gfx1250) — matrix operand order repaired and checked by reading, not compiled, not run on hardware
//
#include <hip/hip_runtime.h>
#include <math.h>

// ---- problem constants (match reference) ----
#define BATCH   2
#define SEQ     2048
#define D_MODEL 512
#define NHEADS  8
#define HD      64
#define HALF_HD 32
#define RATIO   8
#define STRIDE  4
#define WINDOW  128
#define LC      511   // (SEQ - RATIO) / STRIDE + 1

typedef _Float16 v16h __attribute__((ext_vector_type(16)));
typedef _Float16 v8h  __attribute__((ext_vector_type(8)));
typedef float    v8f  __attribute__((ext_vector_type(8)));
typedef float    v4f  __attribute__((ext_vector_type(4)));

__device__ __forceinline__ v8f wmma_f16(v16h a, v16h b, v8f c) {
  // D = A(16x32 f16) * B(32x16 f16) + C(16x16 f32)
  return __builtin_amdgcn_wmma_f32_16x16x32_f16(
      /*neg_a=*/false, a, /*neg_b=*/false, b,
      /*c_mod=*/(short)0, c, /*reuse_a=*/false, /*reuse_b=*/false);
}

__device__ __forceinline__ v4f ld4(const float* p) { return *(const v4f*)p; }

// ------------------------------------------------------------------
// gate softmax over 8 logits (single block)
// ------------------------------------------------------------------
__global__ void gate_softmax_kernel(const float* __restrict__ gl,
                                    float* __restrict__ gw) {
  if (threadIdx.x == 0) {
    float m = -1e30f;
    for (int i = 0; i < RATIO; ++i) m = fmaxf(m, gl[i]);
    float e[RATIO], s = 0.f;
    for (int i = 0; i < RATIO; ++i) { e[i] = __expf(gl[i] - m); s += e[i]; }
    for (int i = 0; i < RATIO; ++i) gw[i] = e[i] / s;
  }
}

// ------------------------------------------------------------------
// RoPE cos/sin tables: [SEQ][32]
// ------------------------------------------------------------------
__global__ void rope_table_kernel(float* __restrict__ cosT,
                                  float* __restrict__ sinT) {
  int idx = blockIdx.x * blockDim.x + threadIdx.x;
  if (idx >= SEQ * HALF_HD) return;
  int l = idx / HALF_HD, i = idx % HALF_HD;
  float inv = expf(-(float)i * (logf(10000.0f) / (float)HALF_HD));
  float f = (float)l * inv;
  cosT[idx] = cosf(f);
  sinT[idx] = sinf(f);
}

// ------------------------------------------------------------------
// x_c[b,w,:] = sum_r gate_w[r] * x[b, min(4w+r, L-1), :]
// ------------------------------------------------------------------
__global__ void compress_kernel(const float* __restrict__ x,
                                const float* __restrict__ gw,
                                float* __restrict__ xc) {
  int bw = blockIdx.x;               // 0 .. B*LC-1
  int b = bw / LC, w = bw % LC;
  float g[RATIO];
#pragma unroll
  for (int r = 0; r < RATIO; ++r) g[r] = gw[r];
  for (int d = threadIdx.x; d < D_MODEL; d += blockDim.x) {
    float acc = 0.f;
#pragma unroll
    for (int r = 0; r < RATIO; ++r) {
      int l = w * STRIDE + r; if (l > SEQ - 1) l = SEQ - 1;
      acc += g[r] * x[((size_t)b * SEQ + l) * D_MODEL + d];
    }
    xc[((size_t)b * LC + w) * D_MODEL + d] = acc;
  }
}

// ------------------------------------------------------------------
// WMMA GEMM: Y[m][n] = sum_k A[m][k] * W[n][k], N = K = 512 compile-time.
// Block = 128 threads (4 waves) covering 64(M) x 128(N).
// Each wave: 32x64 tile = 2 A-frags x 4 B-frags -> 8 WMMAs per 32-K chunk.
// All fragment loads are float4 (b128); stores are immediate-offset b32.
// ------------------------------------------------------------------
template <bool EDGE>
__global__ void __launch_bounds__(128)
gemm_xwT_wmma(const float* __restrict__ A, const float* __restrict__ W,
              float* __restrict__ Y, int M) {
  constexpr int N = 512, K = 512;
  const int lane = threadIdx.x & 31;
  const int wave = threadIdx.x >> 5;
  const int half = lane >> 4, l16 = lane & 15;
  const int m0 = blockIdx.x * 64 + (wave >> 1) * 32;
  const int n0 = blockIdx.y * 128 + (wave & 1) * 64;

  const float* Ar[2];
#pragma unroll
  for (int mt = 0; mt < 2; ++mt) {
    int r = m0 + mt * 16 + l16;
    if (EDGE && r > M - 1) r = M - 1;
    Ar[mt] = A + (size_t)r * K;
  }
  const float* Wr[4];
#pragma unroll
  for (int nt = 0; nt < 4; ++nt)
    Wr[nt] = W + (size_t)(n0 + nt * 16 + l16) * K;

  v8f acc[2][4];
  v8f vzero = {};
#pragma unroll
  for (int mt = 0; mt < 2; ++mt)
#pragma unroll
    for (int nt = 0; nt < 4; ++nt) acc[mt][nt] = vzero;

  for (int k0 = 0; k0 < K; k0 += 32) {
    v16h a[2];
#pragma unroll
    for (int mt = 0; mt < 2; ++mt) {
      // A frag: runs [8h, 8h+8) and [16+8h, 16+8h+8) of this K-chunk
      const float* p = Ar[mt] + k0 + 8 * half;
      v4f r0 = ld4(p), r1 = ld4(p + 4), r2 = ld4(p + 16), r3 = ld4(p + 20);
#pragma unroll
      for (int i = 0; i < 4; ++i) {
        a[mt][i]      = (_Float16)r0[i];
        a[mt][4 + i]  = (_Float16)r1[i];
        a[mt][8 + i]  = (_Float16)r2[i];
        a[mt][12 + i] = (_Float16)r3[i];
      }
    }
#pragma unroll
    for (int nt = 0; nt < 4; ++nt) {
      // B frag: runs [8h, 8h+8) and [16+8h, 16+8h+8) of this K-chunk
      const float* p = Wr[nt] + k0 + 8 * half;
      v4f r0 = ld4(p), r1 = ld4(p + 4), r2 = ld4(p + 16), r3 = ld4(p + 20);
      v16h b;
#pragma unroll
      for (int i = 0; i < 4; ++i) {
        b[i]      = (_Float16)r0[i];
        b[4 + i]  = (_Float16)r1[i];
        b[8 + i]  = (_Float16)r2[i];
        b[12 + i] = (_Float16)r3[i];
      }
#pragma unroll
      for (int mt = 0; mt < 2; ++mt)
        acc[mt][nt] = wmma_f16(a[mt], b, acc[mt][nt]);
    }
  }

#pragma unroll
  for (int mt = 0; mt < 2; ++mt)
#pragma unroll
    for (int nt = 0; nt < 4; ++nt) {
      float* yb = Y + (size_t)(m0 + mt * 16 + 8 * half) * N + n0 + nt * 16 + l16;
#pragma unroll
      for (int v = 0; v < 8; ++v) {
        if (!EDGE || (m0 + mt * 16 + 8 * half + v) < M)
          yb[(size_t)v * N] = acc[mt][nt][v];
      }
    }
}

// ------------------------------------------------------------------
// In-place RoPE on a (B*SEQ, D) buffer viewed as heads of 64
// ------------------------------------------------------------------
__global__ void rope_kernel(float* __restrict__ t,
                            const float* __restrict__ cosT,
                            const float* __restrict__ sinT) {
  int idx = blockIdx.x * blockDim.x + threadIdx.x;   // B*SEQ*H*32 elems
  if (idx >= BATCH * SEQ * NHEADS * HALF_HD) return;
  int i = idx & (HALF_HD - 1);
  int h = (idx / HALF_HD) % NHEADS;
  int l = (idx / (HALF_HD * NHEADS)) % SEQ;
  int b = idx / (HALF_HD * NHEADS * SEQ);
  size_t base = ((size_t)(b * SEQ + l)) * D_MODEL + h * HD;
  float x1 = t[base + i], x2 = t[base + HALF_HD + i];
  float c = cosT[l * HALF_HD + i], s = sinT[l * HALF_HD + i];
  t[base + i]           = x1 * c - x2 * s;
  t[base + HALF_HD + i] = x1 * s + x2 * c;
}

// ------------------------------------------------------------------
// Flash attention: 1 wave per (b, h, 16-query tile).
// Scores via WMMA (q 16x64 f16 x k^T 64x16), online softmax with
// wave32 shfl_xor row reductions, P re-laid C->A via 1KB LDS, PV via WMMA.
// Sink handled by init m=sink, l=1. Fragment loads vectorized (b128).
// ------------------------------------------------------------------
__global__ void __launch_bounds__(32)
attn_kernel(const float* __restrict__ q,  const float* __restrict__ k,
            const float* __restrict__ v,  const float* __restrict__ kc,
            const float* __restrict__ vc, const float* __restrict__ sinkl,
            float* __restrict__ ctx) {
  __shared__ _Float16 pS[16 * 32];   // 16 rows x 32 keys

  int blk = blockIdx.x;
  int qt = blk % (SEQ / 16);
  int h  = (blk / (SEQ / 16)) % NHEADS;
  int b  = blk / ((SEQ / 16) * NHEADS);
  int q0 = qt * 16;

  int lane = threadIdx.x & 31;
  int half = lane >> 4, l16 = lane & 15;

  // q A-fragments, pre-scaled by 1/sqrt(HD)
  const float* qrow = q + ((size_t)(b * SEQ + q0 + l16)) * D_MODEL + h * HD;
  const float qscale = 0.125f;
  v16h qa0, qa1;
  {
    const float* p0 = qrow + 8 * half;
    v4f r0 = ld4(p0), r1 = ld4(p0 + 4), r2 = ld4(p0 + 16), r3 = ld4(p0 + 20);
    const float* p1 = qrow + 32 + 8 * half;
    v4f s0 = ld4(p1), s1 = ld4(p1 + 4), s2 = ld4(p1 + 16), s3 = ld4(p1 + 20);
#pragma unroll
    for (int i = 0; i < 4; ++i) {
      qa0[i]      = (_Float16)(r0[i] * qscale);
      qa0[4 + i]  = (_Float16)(r1[i] * qscale);
      qa0[8 + i]  = (_Float16)(r2[i] * qscale);
      qa0[12 + i] = (_Float16)(r3[i] * qscale);
      qa1[i]      = (_Float16)(s0[i] * qscale);
      qa1[4 + i]  = (_Float16)(s1[i] * qscale);
      qa1[8 + i]  = (_Float16)(s2[i] * qscale);
      qa1[12 + i] = (_Float16)(s3[i] * qscale);
    }
  }

  float sink = sinkl[h];
  float mrow[8], lrow[8];
  v8f acc[4];
  v8f vzero = {};
#pragma unroll
  for (int t = 0; t < 4; ++t) acc[t] = vzero;
#pragma unroll
  for (int vv = 0; vv < 8; ++vv) { mrow[vv] = sink; lrow[vv] = 1.0f; }

  auto processChunk = [&](const float* kbuf, const float* vbuf, int kb,
                          int Ltot, bool comp) {
    float chunkmax[8];
#pragma unroll
    for (int vv = 0; vv < 8; ++vv) chunkmax[vv] = -INFINITY;

    v8f sc[2];
#pragma unroll
    for (int s2 = 0; s2 < 2; ++s2) {
      int key = kb + 16 * s2 + l16;                     // this lane's key column
      int keyc = key < 0 ? 0 : (key > Ltot - 1 ? Ltot - 1 : key);
      const float* krow = kbuf + ((size_t)(b * Ltot + keyc)) * D_MODEL + h * HD;
      // B frag (K = head dim): per 32-chunk, runs [8h,8h+8) and [16+8h,16+8h+8)
      v16h bb0, bb1;
      {
        const float* p0 = krow + 8 * half;
        v4f r0 = ld4(p0), r1 = ld4(p0 + 4), r2 = ld4(p0 + 16), r3 = ld4(p0 + 20);
        const float* p1 = krow + 32 + 8 * half;
        v4f t0 = ld4(p1), t1 = ld4(p1 + 4), t2 = ld4(p1 + 16), t3 = ld4(p1 + 20);
#pragma unroll
        for (int i = 0; i < 4; ++i) {
          bb0[i]      = (_Float16)r0[i];
          bb0[4 + i]  = (_Float16)r1[i];
          bb0[8 + i]  = (_Float16)r2[i];
          bb0[12 + i] = (_Float16)r3[i];
          bb1[i]      = (_Float16)t0[i];
          bb1[4 + i]  = (_Float16)t1[i];
          bb1[8 + i]  = (_Float16)t2[i];
          bb1[12 + i] = (_Float16)t3[i];
        }
      }
      v8f sv = vzero;
      sv = wmma_f16(qa0, bb0, sv);
      sv = wmma_f16(qa1, bb1, sv);
#pragma unroll
      for (int vv = 0; vv < 8; ++vv) {
        int qpos = q0 + vv + 8 * half;
        bool valid;
        if (comp) valid = (key < Ltot) && (qpos >= key * STRIDE + (RATIO - 1));
        else      valid = (key >= 0) && (key < Ltot) &&
                          (qpos - key >= 0) && (qpos - key < WINDOW);
        float val = valid ? sv[vv] : -INFINITY;
        sv[vv] = val;
        chunkmax[vv] = fmaxf(chunkmax[vv], val);
      }
      sc[s2] = sv;
    }
    // row max across the 16 lanes holding this row (stays within each half)
#pragma unroll
    for (int vv = 0; vv < 8; ++vv) {
      chunkmax[vv] = fmaxf(chunkmax[vv], __shfl_xor(chunkmax[vv], 1, 32));
      chunkmax[vv] = fmaxf(chunkmax[vv], __shfl_xor(chunkmax[vv], 2, 32));
      chunkmax[vv] = fmaxf(chunkmax[vv], __shfl_xor(chunkmax[vv], 4, 32));
      chunkmax[vv] = fmaxf(chunkmax[vv], __shfl_xor(chunkmax[vv], 8, 32));
    }
    // merge running stats; rescale accumulators
    float alpha[8];
#pragma unroll
    for (int vv = 0; vv < 8; ++vv) {
      float mnew = fmaxf(mrow[vv], chunkmax[vv]);   // mrow finite (>= sink)
      alpha[vv] = __expf(mrow[vv] - mnew);
      mrow[vv] = mnew;
      lrow[vv] *= alpha[vv];
    }
#pragma unroll
    for (int t = 0; t < 4; ++t)
#pragma unroll
      for (int vv = 0; vv < 8; ++vv) acc[t][vv] *= alpha[vv];

    // p = exp(s - m)  (masked s = -inf -> p = 0), write to LDS, sum rows
    float lsum[8];
#pragma unroll
    for (int vv = 0; vv < 8; ++vv) lsum[vv] = 0.f;
#pragma unroll
    for (int s2 = 0; s2 < 2; ++s2)
#pragma unroll
      for (int vv = 0; vv < 8; ++vv) {
        float p = __expf(sc[s2][vv] - mrow[vv]);
        lsum[vv] += p;
        pS[(vv + 8 * half) * 32 + 16 * s2 + l16] = (_Float16)p;
      }
#pragma unroll
    for (int vv = 0; vv < 8; ++vv) {
      lsum[vv] += __shfl_xor(lsum[vv], 1, 32);
      lsum[vv] += __shfl_xor(lsum[vv], 2, 32);
      lsum[vv] += __shfl_xor(lsum[vv], 4, 32);
      lsum[vv] += __shfl_xor(lsum[vv], 8, 32);
      lrow[vv] += lsum[vv];
    }
    __syncthreads();
    // re-read P in A-fragment layout: two b128 LDS loads
    v16h pa;
    {
      v8h p0 = *(const v8h*)(pS + l16 * 32 + 16 * half);
      v8h p1 = *(const v8h*)(pS + l16 * 32 + 16 * half + 8);
#pragma unroll
      for (int i = 0; i < 8; ++i) { pa[i] = p0[i]; pa[8 + i] = p1[i]; }
    }
    __syncthreads();
    // PV: B frag = V[32 keys x 16 dims]; masked keys have p==0 exactly
#pragma unroll
    for (int t = 0; t < 4; ++t) {
      v16h vb;
#pragma unroll
      for (int j = 0; j < 8; ++j) {
        int k1 = kb + 16 * half + 2 * j;
        int k2 = k1 + 1;
        int k1c = k1 < 0 ? 0 : (k1 > Ltot - 1 ? Ltot - 1 : k1);
        int k2c = k2 < 0 ? 0 : (k2 > Ltot - 1 ? Ltot - 1 : k2);
        vb[2 * j]     = (_Float16)vbuf[((size_t)(b * Ltot + k1c)) * D_MODEL +
                                       h * HD + t * 16 + l16];
        vb[2 * j + 1] = (_Float16)vbuf[((size_t)(b * Ltot + k2c)) * D_MODEL +
                                       h * HD + t * 16 + l16];
      }
      acc[t] = wmma_f16(pa, vb, acc[t]);
    }
  };

  // compressed keys: valid w <= (qpos - 7)/4; tile max qpos = q0+15
  int wmax = (q0 + 8) / STRIDE;
  if (wmax > LC - 1) wmax = LC - 1;
  for (int kb = 0; kb <= wmax; kb += 32)
    processChunk(kc, vc, kb, LC, true);

  // sliding-window keys in [q0-127, q0+15]
  int kb0 = q0 - (WINDOW - 1); if (kb0 < 0) kb0 = 0; kb0 &= ~31;
  for (int kb = kb0; kb <= q0 + 15; kb += 32)
    processChunk(k, v, kb, SEQ, false);

  // finalize (sink already in lrow via init) and write ctx [B*SEQ, D]
  float* cb = ctx + ((size_t)(b * SEQ + q0 + 8 * half)) * D_MODEL + h * HD + l16;
#pragma unroll
  for (int t = 0; t < 4; ++t)
#pragma unroll
    for (int vv = 0; vv < 8; ++vv)
      cb[(size_t)vv * D_MODEL + t * 16] = acc[t][vv] / lrow[vv];
}

// ------------------------------------------------------------------
// host-side orchestration
// ------------------------------------------------------------------
extern "C" void kernel_launch(void* const* d_in, const int* in_sizes, int n_in,
                              void* d_out, int out_size, void* d_ws,
                              size_t ws_size, hipStream_t stream) {
  (void)in_sizes; (void)n_in; (void)out_size; (void)ws_size;
  const float* x    = (const float*)d_in[0];
  const float* wq   = (const float*)d_in[1];
  const float* wk   = (const float*)d_in[2];
  const float* wv   = (const float*)d_in[3];
  const float* wo   = (const float*)d_in[4];
  const float* wk_c = (const float*)d_in[5];
  const float* wv_c = (const float*)d_in[6];
  const float* gate = (const float*)d_in[7];
  const float* sink = (const float*)d_in[8];
  float* out = (float*)d_out;

  // workspace layout (f32), total ~10.1M floats (~40.4 MB)
  float* ws = (float*)d_ws;
  const size_t NQ  = (size_t)BATCH * SEQ * D_MODEL;    // 2,097,152
  const size_t NC  = (size_t)BATCH * LC * D_MODEL;     //   523,264
  float* gw   = ws;                   // 8
  float* cosT = gw + 8;               // SEQ*32
  float* sinT = cosT + SEQ * HALF_HD; // SEQ*32
  float* xc   = sinT + SEQ * HALF_HD; // NC
  float* qb   = xc + NC;              // NQ
  float* kb   = qb + NQ;              // NQ
  float* vb   = kb + NQ;              // NQ
  float* kcb  = vb + NQ;              // NC
  float* vcb  = kcb + NC;             // NC
  float* ctx  = vcb + NC;             // NQ

  gate_softmax_kernel<<<1, 32, 0, stream>>>(gate, gw);
  rope_table_kernel<<<(SEQ * HALF_HD + 255) / 256, 256, 0, stream>>>(cosT, sinT);
  compress_kernel<<<BATCH * LC, 128, 0, stream>>>(x, gw, xc);

  dim3 gq((BATCH * SEQ) / 64, D_MODEL / 128);          // 64 x 4
  dim3 gc((BATCH * LC + 63) / 64, D_MODEL / 128);      // 16 x 4
  gemm_xwT_wmma<false><<<gq, 128, 0, stream>>>(x, wq, qb, BATCH * SEQ);
  gemm_xwT_wmma<false><<<gq, 128, 0, stream>>>(x, wk, kb, BATCH * SEQ);
  gemm_xwT_wmma<false><<<gq, 128, 0, stream>>>(x, wv, vb, BATCH * SEQ);
  gemm_xwT_wmma<true ><<<gc, 128, 0, stream>>>(xc, wk_c, kcb, BATCH * LC);
  gemm_xwT_wmma<true ><<<gc, 128, 0, stream>>>(xc, wv_c, vcb, BATCH * LC);

  int nrope = BATCH * SEQ * NHEADS * HALF_HD;
  rope_kernel<<<(nrope + 255) / 256, 256, 0, stream>>>(qb, cosT, sinT);
  rope_kernel<<<(nrope + 255) / 256, 256, 0, stream>>>(kb, cosT, sinT);

  attn_kernel<<<BATCH * NHEADS * (SEQ / 16), 32, 0, stream>>>(
      qb, kb, vb, kcb, vcb, sink, ctx);

  gemm_xwT_wmma<false><<<gq, 128, 0, stream>>>(ctx, wo, out, BATCH * SEQ);
}
